// PolymerJEPAv2_44246753084191
// MI455X (gfx1250) — compile-verified, matrix operand order checked by reading, not run on hardware
//
#include <hip/hip_runtime.h>
#include <hip/hip_bf16.h>

// ---------------------------------------------------------------------------
// Types for CDNA5 WMMA (gfx1250, wave32)
// ---------------------------------------------------------------------------
typedef __bf16 bf16;
typedef __attribute__((ext_vector_type(8)))  bf16  v8bf;
typedef __attribute__((ext_vector_type(16))) bf16  v16bf;
typedef __attribute__((ext_vector_type(8)))  float v8f;

// Problem constants (from the reference)
#define NN   25600      // N nodes
#define NE   102400     // E edges
#define NSN  51200      // NS subgraph nodes
#define ESN  153600     // ES subgraph edges
#define BSZ  256        // B
#define TT   4          // T
#define PAT  2048       // P = B * PPG
#define HH   512        // H
#define NFNK 133
#define NFEK 14
#define PRWK 16
#define KP_N 160        // 133 padded to mult of 32
#define KP_E 32         // 14  padded
#define KP_P 32         // 16  padded

__device__ __forceinline__ v16bf cat8(v8bf lo, v8bf hi) {
  return __builtin_shufflevector(lo, hi, 0,1,2,3,4,5,6,7,8,9,10,11,12,13,14,15);
}

// ---------------------------------------------------------------------------
// zero fill (graph-capture-safe replacement for hipMemsetAsync)
// ---------------------------------------------------------------------------
__global__ void zero_kernel(float* __restrict__ p, long long n) {
  long long i = (long long)blockIdx.x * blockDim.x + threadIdx.x;
  if (i < n) p[i] = 0.0f;
}

// ---------------------------------------------------------------------------
// Weight convert: W[k][n] f32  ->  Wt[n][kpad] bf16 (zero padded in K)
// ---------------------------------------------------------------------------
__global__ void convw_kernel(const float* __restrict__ W, bf16* __restrict__ Wt,
                             int Kdim, int Kpad, int Nout) {
  long long idx = (long long)blockIdx.x * blockDim.x + threadIdx.x;
  long long tot = (long long)Nout * Kpad;
  if (idx >= tot) return;
  int n = (int)(idx / Kpad);
  int k = (int)(idx % Kpad);
  Wt[idx] = (k < Kdim) ? (bf16)W[(long long)k * Nout + n] : (bf16)0.0f;
}

// ---------------------------------------------------------------------------
// Fused WMMA GEMM:  out = act( [rowscale .*] gather(A) @ W  + addsrc + bias )
//   A      : f32 [*, Kdim], row gathered through gatherA (optional)
//   Wt     : bf16 [Nout][Kpad] (pre-transposed, K zero-padded)
//   block  : 256 thr = 8 waves; tile 64(M) x 128(N); wave tile 16 x 64
//   REQUIRES: M % 64 == 0, Nout % 128 == 0 (true for all call sites)
//   FULLK:  Kdim == Kpad (multiple of 32) -> guard-free vectorized staging
//   emits v_wmma_f32_16x16x32_bf16, ds_load_b128 fragment reads,
//   global_prefetch for next tiles
// ---------------------------------------------------------------------------
template<bool FULLK>
__global__ __launch_bounds__(256)
void gemm_wmma_kernel(const float* __restrict__ A, const int* __restrict__ gatherA,
                      const bf16* __restrict__ Wt, const float* __restrict__ bias,
                      const float* __restrict__ addsrc,
                      const float* __restrict__ rowscale, const int* __restrict__ rsgather,
                      float* __restrict__ outf, bf16* __restrict__ outb,
                      int M, int Kdim, int Kpad, int Nout, int relu)
{
  __shared__ __align__(16) bf16 As[64 * 32];     // 4 KB
  __shared__ __align__(16) bf16 Bs[128 * 32];    // 8 KB

  const int tid  = threadIdx.x;
  const int lane = tid & 31;
  const int wave = tid >> 5;
  const int wm   = wave & 3;          // M sub-tile 0..3
  const int wn   = wave >> 2;         // N sub-tile 0..1
  const int half = lane >> 4;         // K-half select (ISA 16-bit layout)
  const int l16  = lane & 15;

  const int m0 = blockIdx.y * 64;
  const int n0 = blockIdx.x * 128;

  // A staging role: 8 contiguous K per thread
  const int arow = tid >> 2;                 // 0..63
  const int ak0  = (tid & 3) * 8;            // 0,8,16,24
  const int grow = m0 + arow;
  const int asrc = gatherA ? gatherA[grow] : grow;
  float rs = 1.0f;
  if (rowscale) {
    int rr = rsgather ? rsgather[grow] : grow;
    rs = rowscale[rr];
  }
  const float* Arow = A + (long long)asrc * Kdim;

  // B staging role: 16 contiguous K per thread
  const int bc  = tid & 127;                 // column 0..127
  const int bk0 = (tid >> 7) * 16;           // 0 or 16
  const bf16* Brow = Wt + (long long)(n0 + bc) * Kpad;

  v8f acc[4];
  #pragma unroll
  for (int s = 0; s < 4; ++s) acc[s] = (v8f){0,0,0,0,0,0,0,0};

  for (int kt = 0; kt < Kpad; kt += 32) {
    // ---- stage A (f32 -> bf16, apply rowscale) ----
    {
      __align__(16) bf16 tmp[8];
      const int kbase = kt + ak0;
      if (FULLK) {
        // guard-free: Kdim == Kpad, rows fully valid
        const float4 f0 = *(const float4*)(Arow + kbase);
        const float4 f1 = *(const float4*)(Arow + kbase + 4);
        tmp[0] = (bf16)(f0.x * rs); tmp[1] = (bf16)(f0.y * rs);
        tmp[2] = (bf16)(f0.z * rs); tmp[3] = (bf16)(f0.w * rs);
        tmp[4] = (bf16)(f1.x * rs); tmp[5] = (bf16)(f1.y * rs);
        tmp[6] = (bf16)(f1.z * rs); tmp[7] = (bf16)(f1.w * rs);
      } else {
        // branch-free tail handling: clamped address + value select
        #pragma unroll
        for (int j = 0; j < 8; ++j) {
          int k  = kbase + j;
          int kc = k < (Kdim - 1) ? k : (Kdim - 1);
          float v = Arow[kc];                    // unconditional load
          v = (k < Kdim) ? v : 0.0f;             // v_cndmask, no exec games
          tmp[j] = (bf16)(v * rs);
        }
      }
      *(v8bf*)&As[arow * 32 + ak0] = *(v8bf*)tmp;
    }
    // ---- stage B (already bf16, contiguous) ----
    {
      v8bf b0 = *(const v8bf*)(Brow + kt + bk0);
      v8bf b1 = *(const v8bf*)(Brow + kt + bk0 + 8);
      *(v8bf*)&Bs[bc * 32 + bk0]     = b0;
      *(v8bf*)&Bs[bc * 32 + bk0 + 8] = b1;
    }
    // prefetch next K tiles into cache while WMMAs run
    if (kt + 32 < Kpad) {
      if (FULLK) __builtin_prefetch(Arow + kt + 32 + ak0, 0, 1);
      __builtin_prefetch(Brow + kt + 32 + bk0, 0, 1);
    }
    __syncthreads();

    // ---- A fragment: row = wm*16 + l16, K split in two 8-runs by half ----
    const int ar = wm * 16 + l16;
    v8bf a_lo = *(const v8bf*)&As[ar * 32 + half * 8];
    v8bf a_hi = *(const v8bf*)&As[ar * 32 + 16 + half * 8];
    v16bf afrag = cat8(a_lo, a_hi);

    // ---- 4 WMMAs over the wave's 16x64 tile ----
    #pragma unroll
    for (int s = 0; s < 4; ++s) {
      const int bcc = wn * 64 + s * 16 + l16;
      v8bf b_lo = *(const v8bf*)&Bs[bcc * 32 + half * 8];
      v8bf b_hi = *(const v8bf*)&Bs[bcc * 32 + 16 + half * 8];
      v16bf bfrag = cat8(b_lo, b_hi);
      acc[s] = __builtin_amdgcn_wmma_f32_16x16x32_bf16(
          false, afrag, false, bfrag, (short)0, acc[s], false, false);
    }
    __syncthreads();
  }

  // ---- epilogue: residual + bias + relu, f32 or bf16 store ----
  #pragma unroll
  for (int s = 0; s < 4; ++s) {
    #pragma unroll
    for (int v = 0; v < 8; ++v) {
      int row = m0 + wm * 16 + half * 8 + v;
      int col = n0 + wn * 64 + s * 16 + l16;
      long long o = (long long)row * Nout + col;
      float val = acc[s][v];
      if (addsrc) val += addsrc[o];
      if (bias)   val += bias[col];
      if (relu)   val = fmaxf(val, 0.0f);
      if (outb) outb[o] = (bf16)val;
      else      outf[o] = val;
    }
  }
}

// ---------------------------------------------------------------------------
// Edge message scatter: msg[dst] += ew * h[src] * e      (e stored bf16)
// one block per edge, 256 threads handle 512 columns
// ---------------------------------------------------------------------------
__global__ __launch_bounds__(256)
void msg_scatter_kernel(const float* __restrict__ h, const bf16* __restrict__ e,
                        const int* __restrict__ eidx, const float* __restrict__ ew,
                        const int* __restrict__ ewg, float* __restrict__ msg, int nE)
{
  int eid = blockIdx.x;
  if (eid >= nE) return;
  int src = eidx[eid];
  int dst = eidx[nE + eid];
  float w = ew[ewg ? ewg[eid] : eid];
  const float* hs = h + (long long)src * HH;
  const bf16*  es = e + (long long)eid * HH;
  float* md = msg + (long long)dst * HH;
  #pragma unroll
  for (int r = 0; r < 2; ++r) {
    int c = threadIdx.x + r * 256;
    atomicAdd(&md[c], w * hs[c] * (float)es[c]);
  }
}

// ---------------------------------------------------------------------------
// Segment sum + count (for segmean); one block per row
// ---------------------------------------------------------------------------
__global__ __launch_bounds__(256)
void seg_add_kernel(const float* __restrict__ v, const int* __restrict__ gather,
                    const int* __restrict__ seg, float* __restrict__ sums,
                    float* __restrict__ cnt, int M)
{
  int row = blockIdx.x;
  if (row >= M) return;
  int s = seg[row];
  const float* src = v + (long long)(gather ? gather[row] : row) * HH;
  float* dst = sums + (long long)s * HH;
  #pragma unroll
  for (int r = 0; r < 2; ++r) {
    int c = threadIdx.x + r * 256;
    atomicAdd(&dst[c], src[c]);
  }
  if (threadIdx.x == 0) atomicAdd(&cnt[s], 1.0f);
}

__global__ void seg_div_kernel(float* __restrict__ sums, const float* __restrict__ cnt, int P)
{
  int s = blockIdx.x;
  if (s >= P) return;
  float d = fmaxf(cnt[s], 1.0f);
  float* p = sums + (long long)s * HH;
  #pragma unroll
  for (int r = 0; r < 2; ++r) p[threadIdx.x + r * 256] /= d;
}

// ---------------------------------------------------------------------------
// Segment max of rw_pos_enc[nmap] into patch_pes (values >= 0, init 0)
// ---------------------------------------------------------------------------
__global__ void segmax_kernel(const float* __restrict__ rw, const int* __restrict__ nmap,
                              const int* __restrict__ seg, float* __restrict__ pmax, int M)
{
  int i = blockIdx.x * blockDim.x + threadIdx.x;
  if (i >= M) return;
  int s = seg[i];
  int n = nmap[i];
  #pragma unroll
  for (int c = 0; c < PRWK; ++c) {
    // int-punned max is correct for non-negative floats
    atomicMax((int*)&pmax[(long long)s * PRWK + c], __float_as_int(rw[(long long)n * PRWK + c]));
  }
}

// ---------------------------------------------------------------------------
// bi = exclusive cumsum(call_n_patches); cidx/tidx builds
// ---------------------------------------------------------------------------
__global__ void build_idx_kernel(const int* __restrict__ cnp, const int* __restrict__ ci,
                                 const int* __restrict__ ti, int* __restrict__ cidx,
                                 int* __restrict__ tidx)
{
  int b = threadIdx.x;
  if (b >= BSZ) return;
  int bi = 0;
  for (int i = 0; i < b; ++i) bi += cnp[i];
  cidx[b] = ci[b] + bi;
  #pragma unroll
  for (int t = 0; t < TT; ++t) tidx[b * TT + t] = ti[b * TT + t] + bi;
}

// ---------------------------------------------------------------------------
// Finalize: target_x | target_y | embeddings.  One block per (b,t).
// out layout: [0,2048) target_x, [2048,4096) target_y, [4096,...) embeddings
// ---------------------------------------------------------------------------
__global__ __launch_bounds__(256)
void finalize_kernel(const float* __restrict__ subx, const float* __restrict__ subfull,
                     const float* __restrict__ pec, const float* __restrict__ pet,
                     const int* __restrict__ cidx, const int* __restrict__ tidx,
                     const float* __restrict__ predw, const float* __restrict__ predb,
                     float* __restrict__ out)
{
  const long long EMB = 4096;
  int bt = blockIdx.x;           // 0..B*T-1
  int b  = bt >> 2;
  int t  = bt & 3;
  int ci = cidx[b];
  int ti = tidx[bt];
  int tid = threadIdx.x;

  __shared__ float sm[256], s0[256], s1[256];
  float pm = 0.0f, p0 = 0.0f, p1 = 0.0f;
  #pragma unroll
  for (int r = 0; r < 2; ++r) {
    int c = tid + r * 256;
    float ctxv = subx[(long long)ci * HH + c] + pec[(long long)b * HH + c];
    float tg   = subfull[(long long)ti * HH + c];
    out[EMB + (long long)(BSZ + bt) * HH + c] = tg;      // embeddings: tgt rows
    if (t == 0) out[EMB + (long long)b * HH + c] = ctxv; // embeddings: ctx rows
    pm += tg;
    float yv = ctxv + pet[(long long)bt * HH + c];
    p0 += yv * predw[c * 2 + 0];
    p1 += yv * predw[c * 2 + 1];
  }
  sm[tid] = pm; s0[tid] = p0; s1[tid] = p1;
  __syncthreads();
  for (int s = 128; s > 0; s >>= 1) {
    if (tid < s) { sm[tid] += sm[tid+s]; s0[tid] += s0[tid+s]; s1[tid] += s1[tid+s]; }
    __syncthreads();
  }
  if (tid == 0) {
    float m = sm[0] / (float)HH;
    out[bt * 2 + 0] = coshf(m);
    out[bt * 2 + 1] = sinhf(m);
    out[2048 + bt * 2 + 0] = s0[0] + predb[0];
    out[2048 + bt * 2 + 1] = s1[0] + predb[1];
  }
}

// ---------------------------------------------------------------------------
// Host orchestration
// ---------------------------------------------------------------------------
extern "C" void kernel_launch(void* const* d_in, const int* in_sizes, int n_in,
                              void* d_out, int out_size, void* d_ws, size_t ws_size,
                              hipStream_t stream)
{
  (void)in_sizes; (void)n_in; (void)out_size; (void)ws_size;

  // Flattened input order: top-level dict insertion order; nested param dicts
  // flattened in jax pytree (alphabetical key) order: We, Wm, Wn, Wo, be, bm, bn, bo
  const float* x     = (const float*)d_in[0];
  const float* eattr = (const float*)d_in[1];
  const float* ew    = (const float*)d_in[2];
  const float* nw    = (const float*)d_in[3];
  const float* rwpe  = (const float*)d_in[4];
  const float* cWe = (const float*)d_in[5];
  const float* cWm = (const float*)d_in[6];
  const float* cWn = (const float*)d_in[7];
  const float* cWo = (const float*)d_in[8];
  const float* cbe = (const float*)d_in[9];
  const float* cbm = (const float*)d_in[10];
  const float* cbn = (const float*)d_in[11];
  const float* cbo = (const float*)d_in[12];
  const float* tWe = (const float*)d_in[13];
  const float* tWm = (const float*)d_in[14];
  const float* tWn = (const float*)d_in[15];
  const float* tWo = (const float*)d_in[16];
  const float* tbe = (const float*)d_in[17];
  const float* tbm = (const float*)d_in[18];
  const float* tbn = (const float*)d_in[19];
  const float* tbo = (const float*)d_in[20];
  const float* pw    = (const float*)d_in[21];
  const float* pb    = (const float*)d_in[22];
  const float* predw = (const float*)d_in[23];
  const float* predb = (const float*)d_in[24];
  const int* eidx   = (const int*)d_in[25];
  const int* nmap   = (const int*)d_in[26];
  const int* csub   = (const int*)d_in[27];
  const int* emap   = (const int*)d_in[28];
  const int* sbatch = (const int*)d_in[29];
  const int* cnp    = (const int*)d_in[30];
  const int* ctxsi  = (const int*)d_in[31];
  const int* tgtsi  = (const int*)d_in[32];

  float* out = (float*)d_out;

  // ---- workspace carve ----
  size_t cur = 0;
  char* base = (char*)d_ws;
  auto carve = [&](size_t bytes) -> void* {
    void* p = base + cur;
    cur = (cur + bytes + 255) & ~(size_t)255;
    return p;
  };
  bf16* wWnC = (bf16*)carve((size_t)HH * KP_N * 2);
  bf16* wWeC = (bf16*)carve((size_t)HH * KP_E * 2);
  bf16* wWmC = (bf16*)carve((size_t)3 * HH * HH * 2);
  bf16* wWoC = (bf16*)carve((size_t)HH * HH * 2);
  bf16* wWnT = (bf16*)carve((size_t)HH * KP_N * 2);
  bf16* wWeT = (bf16*)carve((size_t)HH * KP_E * 2);
  bf16* wWmT = (bf16*)carve((size_t)3 * HH * HH * 2);
  bf16* wWoT = (bf16*)carve((size_t)HH * HH * 2);
  bf16* wPW  = (bf16*)carve((size_t)HH * KP_P * 2);
  float* bufA = (float*)carve((size_t)NSN * HH * 4);
  float* bufB = (float*)carve((size_t)NSN * HH * 4);
  float* bufM = (float*)carve((size_t)NSN * HH * 4);
  bf16*  Ebuf = (bf16*)carve((size_t)ESN * HH * 2);
  float* subx    = (float*)carve((size_t)PAT * HH * 4);
  float* subfull = (float*)carve((size_t)PAT * HH * 4);
  float* cnt     = (float*)carve((size_t)PAT * 4);
  float* ppes    = (float*)carve((size_t)PAT * PRWK * 4);
  float* pencC   = (float*)carve((size_t)BSZ * HH * 4);
  float* pencT   = (float*)carve((size_t)BSZ * TT * HH * 4);
  int*   cidx    = (int*)carve((size_t)BSZ * 4);
  int*   tidx    = (int*)carve((size_t)BSZ * TT * 4);

  auto zero = [&](float* p, long long n) {
    unsigned g = (unsigned)((n + 255) / 256);
    hipLaunchKernelGGL(zero_kernel, dim3(g), dim3(256), 0, stream, p, n);
  };
  auto convw = [&](const float* W, bf16* Wt, int Kdim, int Kpad) {
    long long tot = (long long)HH * Kpad;
    unsigned g = (unsigned)((tot + 255) / 256);
    hipLaunchKernelGGL(convw_kernel, dim3(g), dim3(256), 0, stream, W, Wt, Kdim, Kpad, HH);
  };
  auto gemm = [&](const float* A, const int* gA, const bf16* Wt, const float* bias,
                  const float* addsrc, const float* rsc, const int* rsg,
                  float* of, bf16* ob, int M, int Kdim, int Kpad, int relu) {
    dim3 g(HH / 128, M / 64);
    if (Kdim == Kpad) {
      hipLaunchKernelGGL((gemm_wmma_kernel<true>), g, dim3(256), 0, stream,
                         A, gA, Wt, bias, addsrc, rsc, rsg, of, ob, M, Kdim, Kpad, HH, relu);
    } else {
      hipLaunchKernelGGL((gemm_wmma_kernel<false>), g, dim3(256), 0, stream,
                         A, gA, Wt, bias, addsrc, rsc, rsg, of, ob, M, Kdim, Kpad, HH, relu);
    }
  };

  // ---- 0. weight conversion (f32 [K][512] -> bf16 [512][Kpad]) ----
  convw(cWn, wWnC, NFNK, KP_N);
  convw(cWe, wWeC, NFEK, KP_E);
  for (int t = 0; t < 3; ++t) convw(cWm + (size_t)t * HH * HH, wWmC + (size_t)t * HH * HH, HH, HH);
  convw(cWo, wWoC, HH, HH);
  convw(tWn, wWnT, NFNK, KP_N);
  convw(tWe, wWeT, NFEK, KP_E);
  for (int t = 0; t < 3; ++t) convw(tWm + (size_t)t * HH * HH, wWmT + (size_t)t * HH * HH, HH, HH);
  convw(tWo, wWoT, HH, HH);
  convw(pw, wPW, PRWK, KP_P);

  // ---- 1. context MPNN on subgraphs ----
  // h = relu(x[nmap] @ Wn + bn)
  gemm(x, nmap, wWnC, cbn, nullptr, nullptr, nullptr, bufA, nullptr, NSN, NFNK, KP_N, 1);
  // e = relu(edge_attr[emap] @ We + be)  (stored bf16)
  gemm(eattr, emap, wWeC, cbe, nullptr, nullptr, nullptr, nullptr, Ebuf, ESN, NFEK, KP_E, 1);
  {
    float* hcur = bufA; float* hnext = bufB;
    for (int t = 0; t < 3; ++t) {
      zero(bufM, (long long)NSN * HH);
      hipLaunchKernelGGL(msg_scatter_kernel, dim3(ESN), dim3(256), 0, stream,
                         hcur, Ebuf, csub, ew, emap, bufM, ESN);
      gemm(bufM, nullptr, wWmC + (size_t)t * HH * HH, cbm + (size_t)t * HH,
           hcur, nullptr, nullptr, hnext, nullptr, NSN, HH, HH, 1);
      float* tmp = hcur; hcur = hnext; hnext = tmp;
    }
    // hout = (h * node_weight[nmap]) @ Wo + bo  -> bufM
    gemm(hcur, nullptr, wWoC, cbo, nullptr, nw, nmap, bufM, nullptr, NSN, HH, HH, 0);
  }
  // subgraph_x = segmean(hout, batch_x, P)
  zero(subx, (long long)PAT * HH);
  zero(cnt, PAT);
  hipLaunchKernelGGL(seg_add_kernel, dim3(NSN), dim3(256), 0, stream,
                     bufM, (const int*)nullptr, sbatch, subx, cnt, NSN);
  hipLaunchKernelGGL(seg_div_kernel, dim3(PAT), dim3(256), 0, stream, subx, cnt, PAT);

  // ---- 2. patch position encodings + indices ----
  zero(ppes, (long long)PAT * PRWK);
  hipLaunchKernelGGL(segmax_kernel, dim3((NSN + 255) / 256), dim3(256), 0, stream,
                     rwpe, nmap, sbatch, ppes, NSN);
  hipLaunchKernelGGL(build_idx_kernel, dim3(1), dim3(256), 0, stream,
                     cnp, ctxsi, tgtsi, cidx, tidx);
  // penc_ctx = relu(patch_pes[cidx] @ patch_w + patch_b)   (M=256)
  gemm(ppes, cidx, wPW, pb, nullptr, nullptr, nullptr, pencC, nullptr, BSZ, PRWK, KP_P, 1);
  // penc_tgt = relu(patch_pes[tidx] @ patch_w + patch_b)   (M=1024)
  gemm(ppes, tidx, wPW, pb, nullptr, nullptr, nullptr, pencT, nullptr, BSZ * TT, PRWK, KP_P, 1);

  // ---- 3. target MPNN on full graph (buffers reused) ----
  gemm(x, nullptr, wWnT, tbn, nullptr, nullptr, nullptr, bufA, nullptr, NN, NFNK, KP_N, 1);
  gemm(eattr, nullptr, wWeT, tbe, nullptr, nullptr, nullptr, nullptr, Ebuf, NE, NFEK, KP_E, 1);
  {
    float* hcur = bufA; float* hnext = bufB;
    for (int t = 0; t < 3; ++t) {
      zero(bufM, (long long)NN * HH);
      hipLaunchKernelGGL(msg_scatter_kernel, dim3(NE), dim3(256), 0, stream,
                         hcur, Ebuf, eidx, ew, (const int*)nullptr, bufM, NE);
      gemm(bufM, nullptr, wWmT + (size_t)t * HH * HH, tbm + (size_t)t * HH,
           hcur, nullptr, nullptr, hnext, nullptr, NN, HH, HH, 1);
      float* tmp = hcur; hcur = hnext; hnext = tmp;
    }
    gemm(hcur, nullptr, wWoT, tbo, nullptr, nw, (const int*)nullptr, bufM, nullptr, NN, HH, HH, 0);
  }
  // sub_full = segmean(full[nmap], batch_x, P)
  zero(subfull, (long long)PAT * HH);
  zero(cnt, PAT);
  hipLaunchKernelGGL(seg_add_kernel, dim3(NSN), dim3(256), 0, stream,
                     bufM, nmap, sbatch, subfull, cnt, NSN);
  hipLaunchKernelGGL(seg_div_kernel, dim3(PAT), dim3(256), 0, stream, subfull, cnt, PAT);

  // ---- 4. finalize: target_x, target_y, embeddings ----
  hipLaunchKernelGGL(finalize_kernel, dim3(BSZ * TT), dim3(256), 0, stream,
                     subx, subfull, pencC, pencT, cidx, tidx, predw, predb, out);
}
